// ReversibleTransformerBlock_15977278341165
// MI455X (gfx1250) — matrix operand order repaired and checked by reading, not compiled, not run on hardware
//
#include <hip/hip_runtime.h>
#include <hip/hip_bf16.h>
#include <math.h>

// ---------------------------------------------------------------------------
// MI455X (gfx1250, wave32) transformer block.
// - All GEMMs and attention: v_wmma_f32_16x16x32_f16 (f16 in, f32 acc)
// - GEMM LDS tiles filled by the Tensor Data Mover (tensor_load_to_lds),
//   double-buffered, tracked with TENSORcnt (s_wait_tensorcnt).
// Workspace requirement: ~102 MB of d_ws.
// ---------------------------------------------------------------------------

#define USE_TDM 1

typedef __attribute__((ext_vector_type(16))) _Float16 v16h;
typedef __attribute__((ext_vector_type(8)))  float    v8f;
typedef __attribute__((ext_vector_type(4)))  uint32_t u32x4;
typedef __attribute__((ext_vector_type(8)))  uint32_t u32x8;

#define WMMA16(a, b, c) \
  __builtin_amdgcn_wmma_f32_16x16x32_f16(false, (a), false, (b), (short)0, (c), false, false)

// ---- Tensor Data Mover: 2D f16 tile global -> LDS (ISA 08_async_tensor §8) -
// D# group0: [1:0]=count=1, [63:32]=lds_addr, [120:64]=global_addr, [127:126]=type=2
// D# group1: [17:16]=data_size(1 => 2B), tensor_dim0/1, tile_dim0/1, dim0 stride
static __device__ __forceinline__ void tdm_load_2d(const _Float16* gsrc,
                                                   const _Float16* lds_dst,
                                                   uint32_t rows, uint32_t cols,
                                                   uint64_t row_stride) {
  uint64_t ga = (uint64_t)(uintptr_t)gsrc;
  uint32_t la = (uint32_t)(uintptr_t)lds_dst;  // flat LDS addr[31:0] == LDS byte offset
  const uint32_t td0 = 1u << 20, td1 = 1u << 20;  // generous bounds: tiles never OOB
  u32x4 g0;
  g0[0] = 1u;                                            // count=1, no gather
  g0[1] = la;                                            // lds_addr
  g0[2] = (uint32_t)ga;                                  // global_addr[31:0]
  g0[3] = (uint32_t)((ga >> 32) & 0x01FFFFFFu) | (2u << 30);  // addr[56:32] | type=2
  u32x8 g1;
  g1[0] = 1u << 16;                                      // data_size=1 (2 bytes)
  g1[1] = (td0 & 0xFFFFu) << 16;                         // tensor_dim0[15:0]
  g1[2] = ((td0 >> 16) & 0xFFFFu) | ((td1 & 0xFFFFu) << 16);
  g1[3] = ((td1 >> 16) & 0xFFFFu) | (cols << 16);        // tile_dim0 = cols
  g1[4] = rows;                                          // tile_dim1 = rows, tile_dim2=0
  g1[5] = (uint32_t)row_stride;                          // tensor_dim0_stride[31:0]
  g1[6] = (uint32_t)(row_stride >> 32) & 0xFFFFu;        // stride[47:32]
  g1[7] = 0u;
  asm volatile("tensor_load_to_lds %0, %1" ::"s"(g0), "s"(g1) : "memory");
}

// ---- fragment loaders (contiguous k = 16*(lane/16) + i, wave32) -------------------------

static __device__ __forceinline__ v16h load_a_frag(const _Float16* base, int row0, int k0, int ld) {
  int lane = threadIdx.x & 31;
  int m = lane & 15;
  int kb = (lane < 16) ? 0 : 16;
  const _Float16* p = base + (size_t)(row0 + m) * ld + k0 + kb;
  v16h a;
#pragma unroll
  for (int i = 0; i < 8; ++i) a[i] = p[i];
#pragma unroll
  for (int i = 0; i < 8; ++i) a[8 + i] = p[8 + i];
  return a;
}

static __device__ __forceinline__ v16h load_b_frag_rm(const _Float16* base, int k0, int col0, int ld) {
  int lane = threadIdx.x & 31;
  int n = lane & 15;
  int kb = (lane < 16) ? 0 : 16;
  const _Float16* p = base + (size_t)(k0 + kb) * ld + col0 + n;
  v16h b;
#pragma unroll
  for (int i = 0; i < 16; ++i) b[i] = p[(size_t)i * ld];
  return b;
}

static __device__ __forceinline__ v16h load_b_frag_tr(const _Float16* base, int k0, int col0, int ld) {
  int lane = threadIdx.x & 31;
  int n = lane & 15;
  int kb = (lane < 16) ? 0 : 16;
  const _Float16* p = base + (size_t)(col0 + n) * ld + k0 + kb;
  v16h b;
#pragma unroll
  for (int i = 0; i < 16; ++i) b[i] = p[i];
  return b;
}

// ---- elementwise fp32 -> f16 ----------------------------------------------

__global__ __launch_bounds__(256) void f32_to_f16_kern(const float* __restrict__ in,
                                                       _Float16* __restrict__ out, int n) {
  int i = blockIdx.x * 256 + threadIdx.x;
  if (i < n) out[i] = (_Float16)in[i];
}

// ---- LayerNorm (row of 512) -> f16 ----------------------------------------

__global__ __launch_bounds__(256) void layernorm_to_f16(const float* __restrict__ x,
                                                        const float* __restrict__ g,
                                                        const float* __restrict__ bta,
                                                        _Float16* __restrict__ y) {
  const int D = 512;
  int row = blockIdx.x;
  int tid = threadIdx.x;
  const float* xr = x + (size_t)row * D;
  float v0 = xr[tid], v1 = xr[tid + 256];
  float s = v0 + v1, s2 = v0 * v0 + v1 * v1;
#pragma unroll
  for (int off = 16; off > 0; off >>= 1) {
    s += __shfl_down(s, off, 32);
    s2 += __shfl_down(s2, off, 32);
  }
  __shared__ float rs[8], rs2[8];
  __shared__ float mean_s, rstd_s;
  int wv = tid >> 5, ln = tid & 31;
  if (ln == 0) { rs[wv] = s; rs2[wv] = s2; }
  __syncthreads();
  if (tid == 0) {
    float ts = 0.f, ts2 = 0.f;
#pragma unroll
    for (int i = 0; i < 8; ++i) { ts += rs[i]; ts2 += rs2[i]; }
    float mu = ts * (1.0f / D);
    float var = ts2 * (1.0f / D) - mu * mu;
    mean_s = mu;
    rstd_s = rsqrtf(var + 1e-5f);
  }
  __syncthreads();
  float mu = mean_s, rstd = rstd_s;
  y[(size_t)row * D + tid]       = (_Float16)((v0 - mu) * rstd * g[tid] + bta[tid]);
  y[(size_t)row * D + tid + 256] = (_Float16)((v1 - mu) * rstd * g[tid + 256] + bta[tid + 256]);
}

// ---- WMMA GEMM: C = A[MxK] * B[KxN], TDM double-buffered LDS tiles ---------
// block = 256 threads (8 waves), block tile 128x64, K-step 32.
// waves: 4 in M x 2 in N, each wave computes 32x32 (2x2 16x16 frags).
// Wave 0 drives the TDM: prefetch tile t+1 while all waves compute tile t.
// EPI 0: scatter to q/k/v f16 [B,H,N,DH]   (N must be 1536, M 8192)
// EPI 1: outF = acc + bias[col] + resid[row*N+col]   (fp32)
// EPI 2: outH = gelu_exact(acc + bias[col])          (f16)

#define GBM 128
#define GBN 64
#define GBK 32

template <int EPI>
__global__ __launch_bounds__(256) void gemm_wmma(const _Float16* __restrict__ A,
                                                 const _Float16* __restrict__ Bm,
                                                 const float* __restrict__ bias,
                                                 const float* __restrict__ resid,
                                                 float* __restrict__ outF,
                                                 _Float16* __restrict__ outH,
                                                 int M, int N, int K) {
  __shared__ _Float16 As[2][GBM * GBK];
  __shared__ _Float16 Bs[2][GBK * GBN];
  int tid = threadIdx.x;
  int wave = tid >> 5;
  int lane = tid & 31;
  int bm = blockIdx.y * GBM;
  int bn = blockIdx.x * GBN;
  int wm = (wave >> 1) * 32;  // 0,32,64,96
  int wn = (wave & 1) * 32;   // 0,32

  v8f acc[2][2];
#pragma unroll
  for (int i = 0; i < 2; ++i)
#pragma unroll
    for (int j = 0; j < 2; ++j) acc[i][j] = (v8f){0.f, 0.f, 0.f, 0.f, 0.f, 0.f, 0.f, 0.f};

  int nk = K / GBK;

#if USE_TDM
  if (tid < 32) {  // wave 0: kick off tile 0 DMA
    tdm_load_2d(A + (size_t)bm * K, &As[0][0], GBM, GBK, (uint64_t)K);
    tdm_load_2d(Bm + bn, &Bs[0][0], GBK, GBN, (uint64_t)N);
  }
  for (int t = 0; t < nk; ++t) {
    int cur = t & 1;
    if (tid < 32) {
      if (t + 1 < nk) {  // prefetch next tile into the other buffer
        int k0n = (t + 1) * GBK;
        tdm_load_2d(A + (size_t)bm * K + k0n, &As[1 - cur][0], GBM, GBK, (uint64_t)K);
        tdm_load_2d(Bm + (size_t)k0n * N + bn, &Bs[1 - cur][0], GBK, GBN, (uint64_t)N);
        // TDM completes in-order per wave: <=2 outstanding => tile t landed.
        __builtin_amdgcn_s_wait_tensorcnt((short)2);
      } else {
        __builtin_amdgcn_s_wait_tensorcnt((short)0);
      }
    }
    __syncthreads();  // release compute waves; tile `cur` is resident
#pragma unroll
    for (int i = 0; i < 2; ++i) {
      v16h a = load_a_frag(&As[cur][0], wm + i * 16, 0, GBK);
#pragma unroll
      for (int j = 0; j < 2; ++j) {
        v16h b = load_b_frag_rm(&Bs[cur][0], 0, wn + j * 16, GBN);
        acc[i][j] = WMMA16(a, b, acc[i][j]);
      }
    }
    __syncthreads();  // everyone done with `cur` before TDM refills it (t+2)
  }
#else
  for (int t = 0; t < nk; ++t) {
    int k0 = t * GBK;
    {
      int r = tid >> 1, c0 = (tid & 1) * 16;
      const float4* src = (const float4*)(A + (size_t)(bm + r) * K + k0 + c0);
      float4* dst = (float4*)(&As[0][0] + r * GBK + c0);
      dst[0] = src[0];
      dst[1] = src[1];
    }
    {
      int r = tid >> 3, c0 = (tid & 7) * 8;
      *(float4*)(&Bs[0][0] + r * GBN + c0) =
          *(const float4*)(Bm + (size_t)(k0 + r) * N + bn + c0);
    }
    __syncthreads();
#pragma unroll
    for (int i = 0; i < 2; ++i) {
      v16h a = load_a_frag(&As[0][0], wm + i * 16, 0, GBK);
#pragma unroll
      for (int j = 0; j < 2; ++j) {
        v16h b = load_b_frag_rm(&Bs[0][0], 0, wn + j * 16, GBN);
        acc[i][j] = WMMA16(a, b, acc[i][j]);
      }
    }
    __syncthreads();
  }
#endif

  // epilogue; C/D layout: VGPR r -> row r (lanes 0-15) / row 8+r (lanes 16-31)
  int n = lane & 15;
  int mb = (lane < 16) ? 0 : 8;
#pragma unroll
  for (int i = 0; i < 2; ++i)
#pragma unroll
    for (int j = 0; j < 2; ++j)
#pragma unroll
      for (int r = 0; r < 8; ++r) {
        int row = bm + wm + i * 16 + mb + r;
        int col = bn + wn + j * 16 + n;
        float v = acc[i][j][r];
        if (EPI == 0) {
          int part = col >> 9;
          int hc = col & 511;
          int h = hc >> 6, dh = hc & 63;
          int bb = row >> 11, nn = row & 2047;
          size_t dst = (size_t)part * (8192ull * 512ull) +
                       (((size_t)(bb * 8 + h) * 2048 + nn) * 64 + dh);
          outH[dst] = (_Float16)v;
        } else if (EPI == 1) {
          size_t idx = (size_t)row * N + col;
          outF[idx] = v + bias[col] + resid[idx];
        } else {
          float t = v + bias[col];
          float gl = 0.5f * t * (1.0f + erff(t * 0.70710678118654752f));
          outH[(size_t)row * N + col] = (_Float16)gl;
        }
      }
}

// ---- LSA flash attention ---------------------------------------------------
// grid (N/64, B*H), block 128 (4 waves). Per block: 64 query rows of one head.
// q/k/v layout [B*H, N, 64] f16. Output [B, N, H*64] f16.

__global__ __launch_bounds__(128) void lsa_attention(const _Float16* __restrict__ qh,
                                                     const _Float16* __restrict__ kh,
                                                     const _Float16* __restrict__ vh,
                                                     const float* __restrict__ tptr,
                                                     _Float16* __restrict__ outH) {
  const int NSEQ = 2048;
  __shared__ _Float16 Qs[64 * 64];
  __shared__ _Float16 Ks[64 * 64];
  __shared__ _Float16 Vs[64 * 64];
  __shared__ _Float16 Ps[64 * 64];

  int tid = threadIdx.x;
  int wave = tid >> 5;
  int lane = tid & 31;
  int qblk = blockIdx.x;
  int bh = blockIdx.y;
  float scale = expf(tptr[0]);  // exp(temperature) = DH^-0.5

  {
    const float4* src = (const float4*)(qh + ((size_t)bh * NSEQ + qblk * 64) * 64);
    float4* dst = (float4*)Qs;
    for (int i = tid; i < 512; i += 128) dst[i] = src[i];
  }

  int mrow0 = wave * 16;
  int n_ = lane & 15;
  int mb = (lane < 16) ? 0 : 8;

  v8f O[4];
#pragma unroll
  for (int j = 0; j < 4; ++j) O[j] = (v8f){0.f, 0.f, 0.f, 0.f, 0.f, 0.f, 0.f, 0.f};
  float rm[8], rl[8];
#pragma unroll
  for (int r = 0; r < 8; ++r) { rm[r] = -3.0e38f; rl[r] = 0.f; }

  for (int t = 0; t < NSEQ / 64; ++t) {
    {
      const float4* ks = (const float4*)(kh + ((size_t)bh * NSEQ + t * 64) * 64);
      const float4* vs = (const float4*)(vh + ((size_t)bh * NSEQ + t * 64) * 64);
      float4* kd = (float4*)Ks;
      float4* vd = (float4*)Vs;
      for (int i = tid; i < 512; i += 128) { kd[i] = ks[i]; vd[i] = vs[i]; }
    }
    __syncthreads();

    v16h a0 = load_a_frag(Qs, mrow0, 0, 64);
    v16h a1 = load_a_frag(Qs, mrow0, 32, 64);
    v8f S[4];
#pragma unroll
    for (int j = 0; j < 4; ++j) {
      v16h b0 = load_b_frag_tr(Ks, 0, j * 16, 64);
      v16h b1 = load_b_frag_tr(Ks, 32, j * 16, 64);
      v8f s = (v8f){0.f, 0.f, 0.f, 0.f, 0.f, 0.f, 0.f, 0.f};
      s = WMMA16(a0, b0, s);
      s = WMMA16(a1, b1, s);
      int gcol = t * 64 + j * 16 + n_;
#pragma unroll
      for (int r = 0; r < 8; ++r) {
        float v = s[r] * scale;
        int grow = qblk * 64 + mrow0 + mb + r;
        if (grow == gcol) v = -3.4e38f;  // LSA diagonal mask
        s[r] = v;
      }
      S[j] = s;
    }

    float tmax[8];
#pragma unroll
    for (int r = 0; r < 8; ++r)
      tmax[r] = fmaxf(fmaxf(S[0][r], S[1][r]), fmaxf(S[2][r], S[3][r]));
#pragma unroll
    for (int off = 1; off < 16; off <<= 1)
#pragma unroll
      for (int r = 0; r < 8; ++r) tmax[r] = fmaxf(tmax[r], __shfl_xor(tmax[r], off, 32));

    float corr[8];
#pragma unroll
    for (int r = 0; r < 8; ++r) {
      float mnew = fmaxf(rm[r], tmax[r]);
      corr[r] = __expf(rm[r] - mnew);
      rm[r] = mnew;
    }
    float tsum[8];
#pragma unroll
    for (int r = 0; r < 8; ++r) tsum[r] = 0.f;
#pragma unroll
    for (int j = 0; j < 4; ++j)
#pragma unroll
      for (int r = 0; r < 8; ++r) {
        float p = __expf(S[j][r] - rm[r]);
        S[j][r] = p;
        tsum[r] += p;
      }
#pragma unroll
    for (int off = 1; off < 16; off <<= 1)
#pragma unroll
      for (int r = 0; r < 8; ++r) tsum[r] += __shfl_xor(tsum[r], off, 32);
#pragma unroll
    for (int r = 0; r < 8; ++r) rl[r] = rl[r] * corr[r] + tsum[r];
#pragma unroll
    for (int j = 0; j < 4; ++j)
#pragma unroll
      for (int r = 0; r < 8; ++r) O[j][r] *= corr[r];

#pragma unroll
    for (int j = 0; j < 4; ++j)
#pragma unroll
      for (int r = 0; r < 8; ++r)
        Ps[(mrow0 + mb + r) * 64 + j * 16 + n_] = (_Float16)S[j][r];

    v16h pa0 = load_a_frag(Ps, mrow0, 0, 64);
    v16h pa1 = load_a_frag(Ps, mrow0, 32, 64);
#pragma unroll
    for (int j = 0; j < 4; ++j) {
      v16h b0 = load_b_frag_rm(Vs, 0, j * 16, 64);
      v16h b1 = load_b_frag_rm(Vs, 32, j * 16, 64);
      O[j] = WMMA16(pa0, b0, O[j]);
      O[j] = WMMA16(pa1, b1, O[j]);
    }
    __syncthreads();
  }

  int b = bh >> 3, h = bh & 7;
  float inv[8];
#pragma unroll
  for (int r = 0; r < 8; ++r) inv[r] = 1.0f / rl[r];
#pragma unroll
  for (int j = 0; j < 4; ++j)
#pragma unroll
    for (int r = 0; r < 8; ++r) {
      int grow = qblk * 64 + mrow0 + mb + r;
      size_t dst = (size_t)(b * NSEQ + grow) * 512 + h * 64 + j * 16 + n_;
      outH[dst] = (_Float16)(O[j][r] * inv[r]);
    }
}

// ---------------------------------------------------------------------------

extern "C" void kernel_launch(void* const* d_in, const int* in_sizes, int n_in,
                              void* d_out, int out_size, void* d_ws, size_t ws_size,
                              hipStream_t stream) {
  const float* x     = (const float*)d_in[0];
  const float* ln1_g = (const float*)d_in[1];
  const float* ln1_b = (const float*)d_in[2];
  const float* Wqkv  = (const float*)d_in[3];
  const float* temp  = (const float*)d_in[4];
  const float* Wo    = (const float*)d_in[5];
  const float* bo    = (const float*)d_in[6];
  const float* ln2_g = (const float*)d_in[7];
  const float* ln2_b = (const float*)d_in[8];
  const float* W1    = (const float*)d_in[9];
  const float* b1    = (const float*)d_in[10];
  const float* W2    = (const float*)d_in[11];
  const float* b2    = (const float*)d_in[12];
  float* z = (float*)d_out;

  const int ROWS = 8192, D = 512, T3 = 1536, MLPD = 2048;

  char* cur = (char*)d_ws;
  auto take = [&](size_t bytes) -> char* {
    char* p = cur;
    cur += (bytes + 255) & ~(size_t)255;
    return p;
  };
  _Float16* xn     = (_Float16*)take((size_t)ROWS * D * 2);
  _Float16* Wqkv16 = (_Float16*)take((size_t)D * T3 * 2);
  _Float16* qkv16  = (_Float16*)take((size_t)3 * ROWS * D * 2);
  _Float16* attn16 = (_Float16*)take((size_t)ROWS * D * 2);
  _Float16* Wo16   = (_Float16*)take((size_t)D * D * 2);
  float*    y      = (float*)take((size_t)ROWS * D * 4);
  _Float16* yn     = (_Float16*)take((size_t)ROWS * D * 2);
  _Float16* W1h    = (_Float16*)take((size_t)D * MLPD * 2);
  _Float16* hh     = (_Float16*)take((size_t)ROWS * MLPD * 2);
  _Float16* W2h    = (_Float16*)take((size_t)MLPD * D * 2);

  f32_to_f16_kern<<<(D * T3 + 255) / 256, 256, 0, stream>>>(Wqkv, Wqkv16, D * T3);
  f32_to_f16_kern<<<(D * D + 255) / 256, 256, 0, stream>>>(Wo, Wo16, D * D);
  f32_to_f16_kern<<<(D * MLPD + 255) / 256, 256, 0, stream>>>(W1, W1h, D * MLPD);
  f32_to_f16_kern<<<(MLPD * D + 255) / 256, 256, 0, stream>>>(W2, W2h, MLPD * D);

  layernorm_to_f16<<<ROWS, 256, 0, stream>>>(x, ln1_g, ln1_b, xn);

  gemm_wmma<0><<<dim3(T3 / GBN, ROWS / GBM), 256, 0, stream>>>(
      xn, Wqkv16, nullptr, nullptr, nullptr, qkv16, ROWS, T3, D);

  lsa_attention<<<dim3(2048 / 64, 32), 128, 0, stream>>>(
      qkv16, qkv16 + (size_t)ROWS * D, qkv16 + (size_t)2 * ROWS * D, temp, attn16);

  gemm_wmma<1><<<dim3(D / GBN, ROWS / GBM), 256, 0, stream>>>(
      attn16, Wo16, bo, x, y, nullptr, ROWS, D, D);

  layernorm_to_f16<<<ROWS, 256, 0, stream>>>(y, ln2_g, ln2_b, yn);

  gemm_wmma<2><<<dim3(MLPD / GBN, ROWS / GBM), 256, 0, stream>>>(
      yn, W1h, b1, nullptr, nullptr, hh, ROWS, MLPD, D);

  gemm_wmma<1><<<dim3(D / GBN, ROWS / GBM), 256, 0, stream>>>(
      hh, W2h, b2, y, z, nullptr, ROWS, D, MLPD);

  hipMemcpyAsync(z + (size_t)ROWS * D, x, (size_t)ROWS * D * sizeof(float),
                 hipMemcpyDeviceToDevice, stream);
}
